// SwinTransformerBlock3D_27195732918796
// MI455X (gfx1250) — matrix operand order repaired and checked by reading, not compiled, not run on hardware
//
#include <hip/hip_runtime.h>
#include <hip/hip_bf16.h>

typedef _Float16 h16 __attribute__((ext_vector_type(16)));
typedef _Float16 h2v __attribute__((ext_vector_type(2)));
typedef float f8 __attribute__((ext_vector_type(8)));

#define TDIM   192
#define THEADS 6
#define THD    32
#define TNW    128
#define TNTOK  392
#define TNPAD  416
#define TLTOK  50176
#define THID   768

// ---------------- fragment loaders (CDNA5 wave32 WMMA f16 16x16x32) ----------------
// A-matrix 16x32 per ISA table: VGPR r<4: K=2r,2r+1 (+8*half); r>=4: K=16+2(r-4) (+8*half)
__device__ __forceinline__ h16 load_afrag(const _Float16* base, int half) {
  h16 f;
#pragma unroll
  for (int r = 0; r < 8; ++r) {
    int k0 = ((r < 4) ? 2 * r : 2 * r + 8) + 8 * half;
    h2v p = *(const h2v*)(base + k0);
    f[2 * r] = p[0];
    f[2 * r + 1] = p[1];
  }
  return f;
}
// B-matrix 32x16: same K order as A: VGPR r<4: K=2r,2r+1 (+8*half); r>=4: K=16+2(r-4) (+8*half)
__device__ __forceinline__ h16 load_bfrag(const _Float16* base, int half) {
  h16 f;
#pragma unroll
  for (int r = 0; r < 8; ++r) {
    int k0 = ((r < 4) ? 2 * r : 2 * r + 8) + 8 * half;
    h2v p = *(const h2v*)(base + k0);
    f[2 * r] = p[0];
    f[2 * r + 1] = p[1];
  }
  return f;
}

__device__ __forceinline__ f8 f8zero() {
  f8 z;
#pragma unroll
  for (int i = 0; i < 8; ++i) z[i] = 0.f;
  return z;
}

// ---------------- fp32 -> f16 weight convert ----------------
__global__ __launch_bounds__(256) void f32_to_f16(const float* __restrict__ s,
                                                  _Float16* __restrict__ d, int n) {
  int i = blockIdx.x * 256 + threadIdx.x;
  if (i < n) d[i] = (_Float16)s[i];
}

// ---------------- build rel-pos-bias (+ key-padding) table: rpb[6][416][416] ----------------
__global__ __launch_bounds__(256) void build_rpb(const float* __restrict__ btab,
                                                 float* __restrict__ rpb) {
  int i = blockIdx.x * 256 + threadIdx.x;
  if (i >= THEADS * TNPAD * TNPAD) return;
  int nk = i % TNPAD;
  int nq = (i / TNPAD) % TNPAD;
  int head = i / (TNPAD * TNPAD);
  float v;
  if (nk >= TNTOK) v = -1.0e30f;        // padded key: softmax kill
  else if (nq >= TNTOK) v = 0.f;        // padded q row: unused output
  else {
    int tdq = nq / 49, thq = (nq / 7) % 7, twq = nq % 7;
    int tdk = nk / 49, thk = (nk / 7) % 7, twk = nk % 7;
    int idx = (tdq - tdk + 7) * 169 + (thq - thk + 6) * 13 + (twq - twk + 6);
    v = btab[idx * THEADS + head];
  }
  rpb[i] = v;
}

// ---------------- build shift-mask region codes: codes[128][416] ----------------
__global__ __launch_bounds__(256) void build_codes(unsigned char* __restrict__ codes) {
  int i = blockIdx.x * 256 + threadIdx.x;
  if (i >= TNW * TNPAD) return;
  int n = i % TNPAD;
  int win = i / TNPAD;
  unsigned char code = 0;
  if (n < TNTOK) {
    int wd = win >> 6, wh = (win >> 3) & 7, ww = win & 7;
    int td = n / 49, th = (n / 7) % 7, tw = n % 7;
    int dd = wd * 8 + td, hh = wh * 7 + th, wp = ww * 7 + tw;
    int rd = dd < 8 ? 0 : (dd < 12 ? 1 : 2);
    int rh = hh < 49 ? 0 : (hh < 53 ? 1 : 2);
    int rw = wp < 49 ? 0 : (wp < 53 ? 1 : 2);
    code = (unsigned char)(rd * 9 + rh * 3 + rw);
  }
  codes[i] = code;
}

// ---------------- LayerNorm1 + cyclic shift + window partition ----------------
__global__ __launch_bounds__(256) void ln_shift_partition(const float* __restrict__ x,
                                                          const float* __restrict__ g,
                                                          const float* __restrict__ b,
                                                          _Float16* __restrict__ xw) {
  int wid = blockIdx.x * 8 + (threadIdx.x >> 5);
  int lane = threadIdx.x & 31;
  if (wid >= TNW * TNTOK) return;
  int win = wid / TNTOK, n = wid % TNTOK;
  int td = n / 49, th = (n / 7) % 7, tw = n % 7;
  int wd = win >> 6, wh = (win >> 3) & 7, ww = win & 7;
  int d = (wd * 8 + td + 4) & 15;
  int h = (wh * 7 + th + 3) % 56;
  int w = (ww * 7 + tw + 3) % 56;
  size_t l = ((size_t)d * 56 + h) * 56 + w;
  const float* xr = x + l * TDIM;
  float vals[6], s = 0.f, sq = 0.f;
#pragma unroll
  for (int i = 0; i < 6; ++i) {
    float v = xr[lane + 32 * i];
    vals[i] = v; s += v; sq += v * v;
  }
#pragma unroll
  for (int m = 16; m >= 1; m >>= 1) { s += __shfl_xor(s, m, 32); sq += __shfl_xor(sq, m, 32); }
  float mean = s * (1.0f / TDIM);
  float var = sq * (1.0f / TDIM) - mean * mean;
  float rstd = rsqrtf(var + 1e-5f);
  _Float16* orow = xw + ((size_t)win * TNPAD + n) * TDIM;
#pragma unroll
  for (int i = 0; i < 6; ++i) {
    int c = lane + 32 * i;
    orow[c] = (_Float16)((vals[i] - mean) * rstd * g[c] + b[c]);
  }
}

// ---------------- plain LayerNorm (LN2) ----------------
__global__ __launch_bounds__(256) void ln_plain(const float* __restrict__ x,
                                                const float* __restrict__ g,
                                                const float* __restrict__ b,
                                                _Float16* __restrict__ out) {
  int wid = blockIdx.x * 8 + (threadIdx.x >> 5);
  int lane = threadIdx.x & 31;
  if (wid >= TLTOK) return;
  const float* xr = x + (size_t)wid * TDIM;
  float vals[6], s = 0.f, sq = 0.f;
#pragma unroll
  for (int i = 0; i < 6; ++i) {
    float v = xr[lane + 32 * i];
    vals[i] = v; s += v; sq += v * v;
  }
#pragma unroll
  for (int m = 16; m >= 1; m >>= 1) { s += __shfl_xor(s, m, 32); sq += __shfl_xor(sq, m, 32); }
  float mean = s * (1.0f / TDIM);
  float var = sq * (1.0f / TDIM) - mean * mean;
  float rstd = rsqrtf(var + 1e-5f);
  _Float16* orow = out + (size_t)wid * TDIM;
#pragma unroll
  for (int i = 0; i < 6; ++i) {
    int c = lane + 32 * i;
    orow[c] = (_Float16)((vals[i] - mean) * rstd * g[c] + b[c]);
  }
}

// ---------------- window reverse + un-shift + residual ----------------
__global__ __launch_bounds__(256) void unpart_residual(const float* __restrict__ x,
                                                       const float* __restrict__ proj,
                                                       float* __restrict__ x2) {
  size_t gid = (size_t)blockIdx.x * 256 + threadIdx.x;
  if (gid >= (size_t)TLTOK * TDIM) return;
  int c = (int)(gid % TDIM);
  size_t l = gid / TDIM;
  int w = (int)(l % 56);
  int h = (int)((l / 56) % 56);
  int d = (int)(l / (56 * 56));
  int ds_ = (d - 4 + 16) & 15;
  int hs = (h - 3 + 56) % 56;
  int ws2 = (w - 3 + 56) % 56;
  int win = ((ds_ / 8) << 6) | ((hs / 7) << 3) | (ws2 / 7);
  int n = (ds_ % 8) * 49 + (hs % 7) * 7 + (ws2 % 7);
  x2[gid] = x[gid] + proj[((size_t)win * TNPAD + n) * TDIM + c];
}

// ---------------- tiled WMMA GEMM: out[M][N] = A[M][K] @ B[K][N] + bias ----------------
template <bool GELU_ACT, bool OUT_F32, bool RESID, bool QKV_SCALE>
__global__ __launch_bounds__(128) void gemm_wmma(const _Float16* __restrict__ A,
                                                 const _Float16* __restrict__ Bw,
                                                 const float* __restrict__ bias,
                                                 const float* __restrict__ resid,
                                                 void* __restrict__ Out,
                                                 int M, int N, int K) {
  __shared__ _Float16 At[64][40];
  __shared__ _Float16 Bt[64][40];  // transposed: Bt[n][k]
  const int mb = blockIdx.x * 64;
  const int nb = blockIdx.y * 64;
  const int t = threadIdx.x;
  const int wv = t >> 5;
  const int lane = t & 31;
  const int lhalf = lane >> 4;
  const int l16 = lane & 15;

  f8 acc[4];
#pragma unroll
  for (int c = 0; c < 4; ++c) acc[c] = f8zero();

  const int arow = t >> 1, ac0 = (t & 1) * 16;
  const int bk = t >> 2, bn0 = (t & 3) * 16;

  for (int kb = 0; kb < K; kb += 32) {
    {  // A tile 64x32, 32B per thread
      const uint32_t* s32 = (const uint32_t*)(A + (size_t)(mb + arow) * K + kb + ac0);
      uint32_t* d32 = (uint32_t*)&At[arow][ac0];
#pragma unroll
      for (int i = 0; i < 8; ++i) d32[i] = s32[i];
    }
    {  // B tile 32x64, stored transposed so K pairs are contiguous per column
      const _Float16* src = Bw + (size_t)(kb + bk) * N + nb + bn0;
#pragma unroll
      for (int i = 0; i < 16; ++i) Bt[bn0 + i][bk] = src[i];
    }
    if (kb + 32 < K) {  // global_prefetch_b8 for the next k-tile
      __builtin_prefetch(A + (size_t)(mb + arow) * K + kb + 32 + ac0, 0, 1);
      __builtin_prefetch(Bw + (size_t)(kb + 32 + bk) * N + nb + bn0, 0, 1);
    }
    __syncthreads();
    h16 afrag = load_afrag(&At[wv * 16 + l16][0], lhalf);
#pragma unroll
    for (int c = 0; c < 4; ++c) {
      h16 bfrag = load_bfrag(&Bt[c * 16 + l16][0], lhalf);
      acc[c] = __builtin_amdgcn_wmma_f32_16x16x32_f16(false, afrag, false, bfrag,
                                                      (short)0, acc[c], false, false);
    }
    __syncthreads();
  }

#pragma unroll
  for (int c = 0; c < 4; ++c) {
#pragma unroll
    for (int j = 0; j < 8; ++j) {
      int row = mb + wv * 16 + j + 8 * lhalf;
      int col = nb + c * 16 + l16;
      float v = acc[c][j] + bias[col];
      if (QKV_SCALE) { if (col < TDIM) v *= 0.17677669529663687f; }  // HD^-0.5
      if (GELU_ACT) v = 0.5f * v * (1.0f + erff(v * 0.7071067811865476f));
      if (RESID) v += resid[(size_t)row * N + col];
      if (OUT_F32) ((float*)Out)[(size_t)row * N + col] = v;
      else         ((_Float16*)Out)[(size_t)row * N + col] = (_Float16)v;
    }
  }
}

// ---------------- flash-style window attention ----------------
// block = 8 waves sharing one (window, head); K/V staged cooperatively in LDS.
// wave = one 16-row q tile; HD=32 == WMMA K.
__global__ __launch_bounds__(256) void attn_kernel(const _Float16* __restrict__ qkv,
                                                   const float* __restrict__ rpb,
                                                   const unsigned char* __restrict__ codes,
                                                   _Float16* __restrict__ out) {
  __shared__ _Float16 Pl[8][16][32];  // per-wave P repack buffer (D-layout -> A-layout)
  __shared__ _Float16 Kl[32][36];     // K block: [key][d]
  __shared__ _Float16 Vt[32][36];     // V block transposed: [d][key]
  const int t = threadIdx.x;
  const int wv = t >> 5;
  const int lane = t & 31;
  const int lhalf = lane >> 4, l16 = lane & 15;
  int blk = blockIdx.x;  // NW*HEADS*4 blocks
  int tgrp = blk & 3; blk >>= 2;
  int head = blk % THEADS;
  int win = blk / THEADS;
  const int tile = tgrp * 8 + wv;
  const bool active = (tile < TNPAD / 16);  // 26 tiles; wave-uniform

  const size_t rowbase = (size_t)win * TNPAD;
  const float* rpbh = rpb + (size_t)head * TNPAD * TNPAD;
  const unsigned char* codew = codes + win * TNPAD;

  h16 aq;
  int nq[8], codeq[8];
  if (active) {
    aq = load_afrag(qkv + (rowbase + tile * 16 + l16) * 576 + head * THD, lhalf);
#pragma unroll
    for (int j = 0; j < 8; ++j) {
      nq[j] = tile * 16 + j + 8 * lhalf;
      codeq[j] = codew[nq[j]];
    }
  }

  float m_st[8], l_st[8];
#pragma unroll
  for (int j = 0; j < 8; ++j) { m_st[j] = -3.0e38f; l_st[j] = 0.f; }
  f8 o0 = f8zero(), o1 = f8zero();

  // cooperative staging indices: thread -> (key, 4 d-values)
  const int skey = t >> 3, sd0 = (t & 7) * 4;

  for (int kb = 0; kb < TNPAD / 32; ++kb) {  // 13 blocks of 32 keys
    {  // stage K (as-is) and V (transposed) blocks: all 256 threads
      const _Float16* srck = qkv + (rowbase + kb * 32 + skey) * 576 + TDIM + head * THD + sd0;
      *(h2v*)&Kl[skey][sd0] = *(const h2v*)srck;
      *(h2v*)&Kl[skey][sd0 + 2] = *(const h2v*)(srck + 2);
      const _Float16* srcv = srck + TDIM;
#pragma unroll
      for (int i = 0; i < 4; ++i) Vt[sd0 + i][skey] = srcv[i];
      if (kb + 1 < TNPAD / 32)
        __builtin_prefetch(srck + 32 * 576, 0, 1);  // next K/V block
    }
    __syncthreads();

    if (active) {
      int key0 = kb * 32 + l16;
      int key1 = key0 + 16;
      // S = q @ k^T : B[d][key] fragments from Kl rows (pairs along d contiguous)
      h16 bk0 = load_bfrag(&Kl[l16][0], lhalf);
      h16 bk1 = load_bfrag(&Kl[16 + l16][0], lhalf);
      f8 z = f8zero();
      f8 s0 = __builtin_amdgcn_wmma_f32_16x16x32_f16(false, aq, false, bk0, (short)0, z, false, false);
      f8 s1 = __builtin_amdgcn_wmma_f32_16x16x32_f16(false, aq, false, bk1, (short)0, z, false, false);

      int ck0 = codew[key0], ck1 = codew[key1];
#pragma unroll
      for (int j = 0; j < 8; ++j) {
        const float* rrow = rpbh + (size_t)nq[j] * TNPAD;
        float v0 = s0[j] + rrow[key0] + (codeq[j] == ck0 ? 0.f : -100.f);
        float v1 = s1[j] + rrow[key1] + (codeq[j] == ck1 ? 0.f : -100.f);
        // row-wise streaming softmax; rows live on one 16-lane half -> width-16 shuffles
        float bm = fmaxf(v0, v1);
#pragma unroll
        for (int m = 8; m >= 1; m >>= 1) bm = fmaxf(bm, __shfl_xor(bm, m, 16));
        float mnew = fmaxf(m_st[j], bm);
        float alpha = __expf(m_st[j] - mnew);
        float p0 = __expf(v0 - mnew);
        float p1 = __expf(v1 - mnew);
        float rs = p0 + p1;
#pragma unroll
        for (int m = 8; m >= 1; m >>= 1) rs += __shfl_xor(rs, m, 16);
        l_st[j] = l_st[j] * alpha + rs;
        m_st[j] = mnew;
        o0[j] *= alpha;
        o1[j] *= alpha;
        Pl[wv][j + 8 * lhalf][l16] = (_Float16)p0;
        Pl[wv][j + 8 * lhalf][16 + l16] = (_Float16)p1;
      }
      // same-wave LDS ops are in-order; just stop the compiler from reordering
      asm volatile("" ::: "memory");
      h16 ap = load_afrag(&Pl[wv][l16][0], lhalf);

      // O += P @ V : B[key][d] fragments from Vt rows (pairs along key contiguous)
      h16 bv0 = load_bfrag(&Vt[l16][0], lhalf);
      h16 bv1 = load_bfrag(&Vt[16 + l16][0], lhalf);
      o0 = __builtin_amdgcn_wmma_f32_16x16x32_f16(false, ap, false, bv0, (short)0, o0, false, false);
      o1 = __builtin_amdgcn_wmma_f32_16x16x32_f16(false, ap, false, bv1, (short)0, o1, false, false);
    }
    __syncthreads();
  }

  if (active) {
#pragma unroll
    for (int j = 0; j < 8; ++j) {
      float rl = (l_st[j] > 0.f) ? (1.0f / l_st[j]) : 0.f;
      int row = tile * 16 + j + 8 * lhalf;
      _Float16* orow = out + (rowbase + row) * TDIM + head * THD;
      orow[l16] = (_Float16)(o0[j] * rl);
      orow[16 + l16] = (_Float16)(o1[j] * rl);
    }
  }
}

// ---------------- host-side launch ----------------
extern "C" void kernel_launch(void* const* d_in, const int* in_sizes, int n_in,
                              void* d_out, int out_size, void* d_ws, size_t ws_size,
                              hipStream_t stream) {
  (void)in_sizes; (void)n_in; (void)out_size; (void)ws_size;
  const float* x        = (const float*)d_in[0];
  const float* gamma1   = (const float*)d_in[1];
  const float* beta1    = (const float*)d_in[2];
  const float* qkv_w    = (const float*)d_in[3];
  const float* qkv_b    = (const float*)d_in[4];
  const float* proj_w   = (const float*)d_in[5];
  const float* proj_b   = (const float*)d_in[6];
  const float* btab     = (const float*)d_in[7];
  const float* gamma2   = (const float*)d_in[8];
  const float* beta2    = (const float*)d_in[9];
  const float* fc1_w    = (const float*)d_in[10];
  const float* fc1_b    = (const float*)d_in[11];
  const float* fc2_w    = (const float*)d_in[12];
  const float* fc2_b    = (const float*)d_in[13];

  const int MROWS = TNW * TNPAD;  // 53248 (= 64*832)
  char* w = (char*)d_ws;
  size_t off = 0;
  auto take = [&](size_t bytes) -> void* {
    void* p = w + off;
    off = (off + bytes + 255) & ~(size_t)255;
    return p;
  };
  _Float16* qkv_w_h  = (_Float16*)take((size_t)TDIM * 3 * TDIM * 2);
  _Float16* proj_w_h = (_Float16*)take((size_t)TDIM * TDIM * 2);
  _Float16* fc1_w_h  = (_Float16*)take((size_t)TDIM * THID * 2);
  _Float16* fc2_w_h  = (_Float16*)take((size_t)THID * TDIM * 2);
  float*    rpbfull  = (float*)take((size_t)THEADS * TNPAD * TNPAD * 4);
  unsigned char* codes = (unsigned char*)take((size_t)TNW * TNPAD);
  _Float16* xw       = (_Float16*)take((size_t)MROWS * TDIM * 2);     // also attn_out (alias)
  _Float16* qkvb     = (_Float16*)take((size_t)MROWS * 3 * TDIM * 2); // also proj_out f32 (alias)
  float*    x2       = (float*)take((size_t)TLTOK * TDIM * 4);
  _Float16* ln2h     = (_Float16*)take((size_t)TLTOK * TDIM * 2);
  _Float16* h1       = (_Float16*)take((size_t)TLTOK * THID * 2);
  _Float16* attn_out = xw;
  float*    proj_out = (float*)qkvb;

  // weights fp32 -> f16; bias/mask lookup tables
  f32_to_f16<<<(TDIM * 3 * TDIM + 255) / 256, 256, 0, stream>>>(qkv_w, qkv_w_h, TDIM * 3 * TDIM);
  f32_to_f16<<<(TDIM * TDIM + 255) / 256, 256, 0, stream>>>(proj_w, proj_w_h, TDIM * TDIM);
  f32_to_f16<<<(TDIM * THID + 255) / 256, 256, 0, stream>>>(fc1_w, fc1_w_h, TDIM * THID);
  f32_to_f16<<<(THID * TDIM + 255) / 256, 256, 0, stream>>>(fc2_w, fc2_w_h, THID * TDIM);
  build_rpb<<<(THEADS * TNPAD * TNPAD + 255) / 256, 256, 0, stream>>>(btab, rpbfull);
  build_codes<<<(TNW * TNPAD + 255) / 256, 256, 0, stream>>>(codes);

  // zero padded window rows, then LN1 + shift + partition
  hipMemsetAsync(xw, 0, (size_t)MROWS * TDIM * 2, stream);
  ln_shift_partition<<<(TNW * TNTOK) / 8, 256, 0, stream>>>(x, gamma1, beta1, xw);

  // QKV GEMM (q scaled by HD^-0.5 in epilogue)
  gemm_wmma<false, false, false, true><<<dim3(MROWS / 64, (3 * TDIM) / 64), 128, 0, stream>>>(
      xw, qkv_w_h, qkv_b, nullptr, qkvb, MROWS, 3 * TDIM, TDIM);

  // window attention (writes attn_out = xw alias)
  attn_kernel<<<TNW * THEADS * 4, 256, 0, stream>>>(qkvb, rpbfull, codes, attn_out);

  // proj GEMM -> fp32 (proj_out aliases qkv region, which is dead now)
  gemm_wmma<false, true, false, false><<<dim3(MROWS / 64, TDIM / 64), 128, 0, stream>>>(
      attn_out, proj_w_h, proj_b, nullptr, proj_out, MROWS, TDIM, TDIM);

  // reverse windows + unshift + residual -> x2
  unpart_residual<<<((size_t)TLTOK * TDIM + 255) / 256, 256, 0, stream>>>(x, proj_out, x2);

  // LN2 -> f16
  ln_plain<<<TLTOK / 8, 256, 0, stream>>>(x2, gamma2, beta2, ln2h);

  // MLP: fc1 + exact GELU -> f16, fc2 + residual -> d_out (fp32)
  gemm_wmma<true, false, false, false><<<dim3(TLTOK / 64, THID / 64), 128, 0, stream>>>(
      ln2h, fc1_w_h, fc1_b, nullptr, h1, TLTOK, THID, TDIM);
  gemm_wmma<false, true, true, false><<<dim3(TLTOK / 64, TDIM / 64), 128, 0, stream>>>(
      h1, fc2_w_h, fc2_b, x2, d_out, TLTOK, TDIM, THID);
}
